// Note_Decoder_43387759624364
// MI455X (gfx1250) — matrix operand order repaired and checked by reading, not compiled, not run on hardware
//
#include <hip/hip_runtime.h>
#include <hip/hip_bf16.h>

#define TQ 1024
#define BB 2048
#define HH 512
#define CC 9
#define GATE 1536   // 3H
#define BT 16       // batch rows per block

typedef __bf16 bf16_t;
typedef __attribute__((ext_vector_type(16))) bf16_t v16bf;
typedef __attribute__((ext_vector_type(8)))  float  v8f;

__device__ __forceinline__ unsigned short f2bf(float x) {
  unsigned int u = __float_as_uint(x);
  u += 0x7fffu + ((u >> 16) & 1u);          // round-to-nearest-even
  return (unsigned short)(u >> 16);
}
__device__ __forceinline__ unsigned short f2h(float x) {
  union { _Float16 h; unsigned short s; } c; c.h = (_Float16)x; return c.s;
}
__device__ __forceinline__ float h2f(unsigned short s) {
  union { unsigned short s; _Float16 h; } c; c.s = s; return (float)c.h;
}
__device__ __forceinline__ float sigm(float x) { return 1.0f / (1.0f + __expf(-x)); }
__device__ __forceinline__ float tanh_fast(float x) {
  x = fminf(fmaxf(x, -15.0f), 15.0f);
  float e = __expf(2.0f * x);
  return (e - 1.0f) / (e + 1.0f);
}

union Frag { v16bf v; uint4 q[2]; };

// A fragment: 16x32 bf16 tile from LDS h-state [16][512], per ISA 16-bit A layout:
// lane L: m=L&15, hi=L>>4; VGPR j in 0..3 -> K=k0+hi*8+2j ; j in 4..7 -> +16.
__device__ __forceinline__ v16bf load_a(const unsigned short* h, int k0, int lane) {
  int m = lane & 15, hi = lane >> 4;
  const unsigned short* row = h + m * HH + k0 + hi * 8;
  Frag f;
  f.q[0] = *(const uint4*)(row);        // K pairs 0,2,4,6   (+hi*8)
  f.q[1] = *(const uint4*)(row + 16);   // K pairs 16,18,20,22
  return f.v;
}
// B fragment: 32x16 bf16, B[k][n] = W[n][k] with W row-major [1536][512] bf16:
// lane L: n=L&15, hi=L>>4; VGPR j in 0..3 -> K=k0+hi*8+2j ; j in 4..7 -> +16.
__device__ __forceinline__ v16bf load_b(const unsigned short* w, int n0, int k0, int lane) {
  int n = lane & 15, hi = lane >> 4;
  const unsigned short* col = w + (size_t)(n0 + n) * HH + k0 + hi * 8;
  Frag f;
  f.q[0] = *(const uint4*)(col);
  f.q[1] = *(const uint4*)(col + 16);
  return f.v;
}

// One [16 x 1536] GEMM pass: gout(f16) = h(bf16) @ W^T + bias, K=512.
// 8 waves x 12 n-tiles, k-outer so the A fragment is reused across 12 WMMAs.
__device__ __forceinline__ void gemm_pass(const unsigned short* hlds,
                                          const unsigned short* wbf,
                                          const float* bias,
                                          unsigned short* gout) {
  const int lane = threadIdx.x & 31;
  const int wave = threadIdx.x >> 5;
  v8f acc[12];
#pragma unroll
  for (int i = 0; i < 12; ++i) {
    float bv = bias[(wave + i * 8) * 16 + (lane & 15)];   // C-layout: n fixed per lane
    acc[i] = (v8f){bv, bv, bv, bv, bv, bv, bv, bv};
  }
#pragma unroll 2
  for (int k0 = 0; k0 < HH; k0 += 32) {
    v16bf a = load_a(hlds, k0, lane);
#pragma unroll
    for (int i = 0; i < 12; ++i) {
      v16bf b = load_b(wbf, (wave + i * 8) * 16, k0, lane);
      acc[i] = __builtin_amdgcn_wmma_f32_16x16x32_bf16(
          false, a, false, b, (short)0, acc[i], false, false);
    }
  }
  // C layout: lanes0-15 -> M=v, lanes16-31 -> M=v+8; N = lane&15
  const int n_l = lane & 15, mbase = (lane >> 4) * 8;
#pragma unroll
  for (int i = 0; i < 12; ++i) {
    int n = (wave + i * 8) * 16 + n_l;
#pragma unroll
    for (int v = 0; v < 8; ++v) gout[(mbase + v) * GATE + n] = f2h(acc[i][v]);
  }
}

// ---------------- pre-kernels ----------------

__global__ void init_h1_kernel(const float* __restrict__ z, const float* __restrict__ se,
                               const float* __restrict__ cond, const float* __restrict__ W_h,
                               const float* __restrict__ b_h, float* __restrict__ h1_0) {
  int idx = blockIdx.x * blockDim.x + threadIdx.x;   // 2048*512 exact
  int b = idx >> 9, h = idx & 511;
  const float* wr = W_h + (size_t)h * 272;
  const float* zr = z + (size_t)b * 128;
  const float* sr = se + (size_t)b * 128;
  const float* cr = cond + (size_t)b * 16;
  float s = b_h[h];
  for (int k = 0; k < 128; ++k) s = fmaf(zr[k], wr[k], s);
  for (int k = 0; k < 128; ++k) s = fmaf(sr[k], wr[128 + k], s);
  for (int k = 0; k < 16; ++k)  s = fmaf(cr[k], wr[256 + k], s);
  h1_0[idx] = tanh_fast(s);
}

__global__ void convert_kernel(const float* __restrict__ a, const float* __restrict__ b,
                               const float* __restrict__ c, unsigned short* __restrict__ ab,
                               unsigned short* __restrict__ bb, unsigned short* __restrict__ cb) {
  int idx = blockIdx.x * blockDim.x + threadIdx.x;
  if (idx < GATE * HH) { ab[idx] = f2bf(a[idx]); bb[idx] = f2bf(b[idx]); cb[idx] = f2bf(c[idx]); }
}

// ---------------- persistent decoder ----------------

__global__ __launch_bounds__(256)
void decoder_kernel(const float* __restrict__ skel,      // [B,T,C]
                    const float* __restrict__ rand_seq,  // [T,B,C]
                    const float* __restrict__ w_ih1,     // [1536,18]
                    const float* __restrict__ b_ih1,
                    const float* __restrict__ b_hh1,
                    const float* __restrict__ b_ih2,
                    const float* __restrict__ b_hh2,
                    const float* __restrict__ W_out,     // [9,512]
                    const float* __restrict__ b_out,
                    const float* __restrict__ h1_0,      // [B,512] f32
                    const unsigned short* __restrict__ whh1,  // bf16 [1536,512]
                    const unsigned short* __restrict__ wih2,
                    const unsigned short* __restrict__ whh2,
                    float* __restrict__ out) {           // [B,T,C]
  __shared__ __align__(16) unsigned short h1bf[BT * HH];     // 16 KB bf16
  __shared__ __align__(16) unsigned short h2bf[BT * HH];     // 16 KB bf16
  __shared__ __align__(16) unsigned short gatesA[BT * GATE]; // 48 KB f16
  __shared__ __align__(16) unsigned short gatesB[BT * GATE]; // 48 KB f16
  __shared__ __align__(16) unsigned short tanhb[BT * HH];    // 16 KB f16
  __shared__ float xbuf[BT][20];                             // [out(9) | sk(9)]

  const int tid = threadIdx.x;
  const int b0 = blockIdx.x * BT;

  // fp32 master hidden states: element i -> row r=i>>1, col h=((i&1)<<8)+tid
  float hr1[32], hr2[32];
#pragma unroll
  for (int i = 0; i < 32; ++i) {
    int r = i >> 1, h = ((i & 1) << 8) + tid;
    float v = h1_0[(size_t)(b0 + r) * HH + h];
    hr1[i] = v; hr2[i] = 0.0f;
    h1bf[r * HH + h] = f2bf(v);
  }
  if (tid < BT * CC) xbuf[tid / CC][tid % CC] = 0.0f;   // out0 = 0
  __syncthreads();

#pragma unroll 1
  for (int t = 0; t < TQ; ++t) {
    // ---- phase A: fetch sk_t; gates_hh1 = h1 @ w_hh1^T + b_hh1 ----
    if (tid < BT * CC) {
      int r = tid / CC, c = tid % CC;
      xbuf[r][CC + c] = skel[((size_t)(b0 + r) * TQ + t) * CC + c];
    }
    gemm_pass(h1bf, whh1, b_hh1, gatesA);
    __syncthreads();

    // ---- phase B: GRU1 pointwise (gi1 = x @ w_ih1^T done inline, K=18) ----
#pragma unroll
    for (int r = 0; r < BT; ++r) {
      float xv[18];
#pragma unroll
      for (int k = 0; k < 18; ++k) xv[k] = xbuf[r][k];
#pragma unroll
      for (int half = 0; half < 2; ++half) {
        int i = r * 2 + half;
        int h = (half << 8) + tid;
        float ir = b_ih1[h], iz = b_ih1[HH + h], in_ = b_ih1[2 * HH + h];
        const float* w0 = w_ih1 + (size_t)h * 18;
        const float* w1 = w_ih1 + (size_t)(HH + h) * 18;
        const float* w2 = w_ih1 + (size_t)(2 * HH + h) * 18;
#pragma unroll
        for (int k = 0; k < 18; ++k) {
          ir = fmaf(xv[k], w0[k], ir);
          iz = fmaf(xv[k], w1[k], iz);
          in_ = fmaf(xv[k], w2[k], in_);
        }
        float hrg = h2f(gatesA[r * GATE + h]);
        float hzg = h2f(gatesA[r * GATE + HH + h]);
        float hng = h2f(gatesA[r * GATE + 2 * HH + h]);
        float rg = sigm(ir + hrg), zg = sigm(iz + hzg);
        float nn = tanh_fast(in_ + rg * hng);
        float hn = (1.0f - zg) * nn + zg * hr1[i];
        hr1[i] = hn;
        unsigned short hb = f2bf(hn);
        h1bf[r * HH + h] = hb;
        if (t == 0) { hr2[i] = hn; h2bf[r * HH + h] = hb; }   // hx[1] = hx[0] at i==0
      }
    }
    __syncthreads();

    // ---- phase C: gi2 and gh2 (kept separate; n-gate needs r*gh_n) ----
    gemm_pass(h1bf, wih2, b_ih2, gatesA);
    gemm_pass(h2bf, whh2, b_hh2, gatesB);
    __syncthreads();

    // ---- phase D: GRU2 pointwise + tanh for out_fc ----
#pragma unroll
    for (int i = 0; i < 32; ++i) {
      int r = i >> 1, h = ((i & 1) << 8) + tid;
      float ar = h2f(gatesA[r * GATE + h]),          br = h2f(gatesB[r * GATE + h]);
      float az = h2f(gatesA[r * GATE + HH + h]),     bz = h2f(gatesB[r * GATE + HH + h]);
      float an = h2f(gatesA[r * GATE + 2 * HH + h]), bn = h2f(gatesB[r * GATE + 2 * HH + h]);
      float rg = sigm(ar + br), zg = sigm(az + bz);
      float nn = tanh_fast(an + rg * bn);
      float hn = (1.0f - zg) * nn + zg * hr2[i];
      hr2[i] = hn;
      h2bf[r * HH + h] = f2bf(hn);
      tanhb[r * HH + h] = f2h(tanh_fast(hn));
    }
    __syncthreads();

    // ---- phase E: note = tanh(h2) @ W_out^T + b_out; Bernoulli sample ----
    if (tid < BT * CC) {
      int r = tid / CC, c = tid % CC;
      const float* wr = W_out + (size_t)c * HH;
      const unsigned short* th = tanhb + r * HH;
      float s = b_out[c];
#pragma unroll 8
      for (int k = 0; k < HH; ++k) s = fmaf(h2f(th[k]), wr[k], s);
      out[((size_t)(b0 + r) * TQ + t) * CC + c] = s;
      float rnd = rand_seq[((size_t)t * BB + b0 + r) * CC + c];
      xbuf[r][c] = (sigm(s) - rnd > 0.0f) ? 1.0f : 0.0f;
    }
    __syncthreads();
  }
}

extern "C" void kernel_launch(void* const* d_in, const int* in_sizes, int n_in,
                              void* d_out, int out_size, void* d_ws, size_t ws_size,
                              hipStream_t stream) {
  const float* z     = (const float*)d_in[0];
  const float* se    = (const float*)d_in[1];
  const float* skel  = (const float*)d_in[2];
  const float* cond  = (const float*)d_in[3];
  const float* rnd   = (const float*)d_in[4];
  const float* W_h   = (const float*)d_in[5];
  const float* b_h   = (const float*)d_in[6];
  const float* w_ih1 = (const float*)d_in[7];
  const float* w_hh1 = (const float*)d_in[8];
  const float* b_ih1 = (const float*)d_in[9];
  const float* b_hh1 = (const float*)d_in[10];
  const float* w_ih2 = (const float*)d_in[11];
  const float* w_hh2 = (const float*)d_in[12];
  const float* b_ih2 = (const float*)d_in[13];
  const float* b_hh2 = (const float*)d_in[14];
  const float* W_out = (const float*)d_in[15];
  const float* b_out = (const float*)d_in[16];
  float* out = (float*)d_out;

  char* ws = (char*)d_ws;
  float*          h1_0 = (float*)(ws + 0);                       // 2048*512*4 = 4 MB
  unsigned short* w1b  = (unsigned short*)(ws + 4194304);        // 1.5 MB bf16
  unsigned short* w2b  = (unsigned short*)(ws + 4194304 + 1572864);
  unsigned short* w3b  = (unsigned short*)(ws + 4194304 + 2 * 1572864);

  init_h1_kernel<<<(BB * HH) / 256, 256, 0, stream>>>(z, se, cond, W_h, b_h, h1_0);
  convert_kernel<<<(GATE * HH + 255) / 256, 256, 0, stream>>>(w_hh1, w_ih2, w_hh2,
                                                              w1b, w2b, w3b);
  decoder_kernel<<<BB / BT, 256, 0, stream>>>(skel, rnd, w_ih1, b_ih1, b_hh1,
                                              b_ih2, b_hh2, W_out, b_out, h1_0,
                                              w1b, w2b, w3b, out);
}
